// EncoderLayer_9878424780808
// MI455X (gfx1250) — matrix operand order repaired and checked by reading, not compiled, not run on hardware
//
#include <hip/hip_runtime.h>
#include <hip/hip_bf16.h>
#include <math.h>

// ---------------------------------------------------------------------------
// MI455X (gfx1250) transformer encoder layer.
// Matmuls: v_wmma_f32_16x16x32_f16 (wave32 WMMA, f32 accum).
// GEMM tile staging: global_load_async_to_lds_b128 + s_wait_asynccnt,
// double-buffered LDS (2-stage pipeline).
// ---------------------------------------------------------------------------

typedef __attribute__((ext_vector_type(16))) _Float16 v16h;
typedef __attribute__((ext_vector_type(8)))  _Float16 v8h;
typedef __attribute__((ext_vector_type(8)))  float    v8f;

#define D_MODEL 1024
#define NHEAD   16
#define HEAD_DIM 64
#define D_FF    4096
#define SEQ     2048
#define BATCH   4
#define M_TOT   (BATCH * SEQ)   // 8192 rows

// Low 32 bits of a flat pointer into the LDS aperture are the wave-relative
// LDS byte offset (CDNA5 ISA 10.2).
__device__ __forceinline__ unsigned lds_addr_of(const void* p) {
    return (unsigned)(uintptr_t)p;
}

// ASYNCcnt-tracked DMA of 16B from global memory into LDS (no VGPR roundtrip).
__device__ __forceinline__ void async_ld_b128(unsigned lds, const void* g) {
    asm volatile("global_load_async_to_lds_b128 %0, %1, off"
                 :: "v"(lds), "v"(g) : "memory");
}
__device__ __forceinline__ void wait_async0() {
    asm volatile("s_wait_asynccnt 0" ::: "memory");
}

// ---------------------------------------------------------------------------
// fp32 [K][N] -> f16 [N][K] transpose+convert (weights)
// grid: (N/32, K/32), block: (32, 8)
// ---------------------------------------------------------------------------
__global__ __launch_bounds__(256)
void transpose_cvt(const float* __restrict__ in, _Float16* __restrict__ out,
                   int K, int N) {
    __shared__ float tile[32][33];
    int tx = threadIdx.x, ty = threadIdx.y;
    int n0 = blockIdx.x * 32, k0 = blockIdx.y * 32;
#pragma unroll
    for (int r = 0; r < 4; ++r) {
        int kk = k0 + ty + r * 8;
        tile[ty + r * 8][tx] = in[(size_t)kk * N + n0 + tx];
    }
    __syncthreads();
#pragma unroll
    for (int r = 0; r < 4; ++r) {
        int nn = n0 + ty + r * 8;
        out[(size_t)nn * K + k0 + tx] = (_Float16)tile[tx][ty + r * 8];
    }
}

// ---------------------------------------------------------------------------
// RMSNorm: fp32 row -> f16 row.  One block (256 thr) per row.
// ---------------------------------------------------------------------------
__global__ __launch_bounds__(256)
void rmsnorm_kernel(const float* __restrict__ x, const float* __restrict__ w,
                    _Float16* __restrict__ out, int D) {
    int row = blockIdx.x;
    const float* xr = x + (size_t)row * D;
    float ss = 0.f;
    for (int i = threadIdx.x; i < D; i += 256) { float v = xr[i]; ss += v * v; }
#pragma unroll
    for (int m = 16; m >= 1; m >>= 1) ss += __shfl_xor(ss, m, 32);
    __shared__ float red[8];
    if ((threadIdx.x & 31) == 0) red[threadIdx.x >> 5] = ss;
    __syncthreads();
    float tot = 0.f;
#pragma unroll
    for (int j = 0; j < 8; ++j) tot += red[j];
    float rms = sqrtf(tot / (float)D);
    float inv = 1.f / (rms + 1e-8f);
    for (int i = threadIdx.x; i < D; i += 256)
        out[(size_t)row * D + i] = (_Float16)(w[i] * xr[i] * inv);
}

// ---------------------------------------------------------------------------
// WMMA GEMM:  C[M][N] = A_f16[M][K] @ BT_f16[N][K]^T  (+ epilogue)
// Block tile 128x128, 8 waves (4 M x 2 N), wave tile 32x64, K-step 64.
// Tiles staged via async global->LDS DMA, double-buffered (2 stages).
// MODE 0: QKV scatter (q scaled 0.125, v stored transposed per head)
// MODE 1: out-proj: +bias +resid -> fp32
// MODE 2: FF1: +bias, exact GELU -> f16
// MODE 3: FF2: +bias +resid -> fp32
// ---------------------------------------------------------------------------
template <int MODE>
__global__ __launch_bounds__(256)
void gemm_f16(const _Float16* __restrict__ A, const _Float16* __restrict__ BT,
              const float* __restrict__ bias, const float* __restrict__ resid,
              float* __restrict__ outF, _Float16* __restrict__ outH,
              _Float16* __restrict__ qO, _Float16* __restrict__ kO,
              _Float16* __restrict__ vTO,
              int M, int N, int K) {
    __shared__ __attribute__((aligned(32))) _Float16 As[2][128 * 64]; // 2x16KB
    __shared__ __attribute__((aligned(32))) _Float16 Bs[2][128 * 64]; // 2x16KB

    const int tid  = threadIdx.x;
    const int lane = tid & 31;
    const int w    = tid >> 5;
    const int wm   = w & 3;        // 4 waves along M
    const int wn   = w >> 2;       // 2 waves along N
    const int lr   = lane & 15;
    const int hs   = lane >> 4;
    const int mBase = blockIdx.y * 128;
    const int nBase = blockIdx.x * 128;

    v8f acc[2][4];
#pragma unroll
    for (int i = 0; i < 2; ++i)
#pragma unroll
        for (int j = 0; j < 4; ++j) acc[i][j] = (v8f){0,0,0,0,0,0,0,0};

    // per-thread DMA mapping: 4 chunks of 16B per matrix per 128x64 tile
    // chunk c = tid*4+u : row = c>>3 (0..127), part = c&7 (8 halves each)
    auto issue_tile = [&](int kt, int st) {
#pragma unroll
        for (int u = 0; u < 4; ++u) {
            int c = tid * 4 + u;
            int row = c >> 3, part = c & 7;
            async_ld_b128(lds_addr_of(&As[st][row * 64 + part * 8]),
                          &A[(size_t)(mBase + row) * K + kt + part * 8]);
            async_ld_b128(lds_addr_of(&Bs[st][row * 64 + part * 8]),
                          &BT[(size_t)(nBase + row) * K + kt + part * 8]);
        }
    };

    const int nkt = K >> 6;        // K / 64 stages
    issue_tile(0, 0);

    for (int t = 0; t < nkt; ++t) {
        wait_async0();             // this wave's DMA into stage t&1 complete
        __syncthreads();           // all waves' DMA visible; prev stage free
        if (t + 1 < nkt) issue_tile((t + 1) << 6, (t + 1) & 1);
        const int st = t & 1;

#pragma unroll
        for (int ks = 0; ks < 2; ++ks) {       // two K=32 sub-steps
            v16h af[2];
#pragma unroll
            for (int i = 0; i < 2; ++i) {
                int r = wm * 32 + i * 16 + lr;
                v8h lo = *(const v8h*)&As[st][r * 64 + ks * 32 + hs * 16];
                v8h hi = *(const v8h*)&As[st][r * 64 + ks * 32 + hs * 16 + 8];
                af[i] = __builtin_shufflevector(lo, hi, 0,1,2,3,4,5,6,7,
                                                         8,9,10,11,12,13,14,15);
            }
            v16h bf[4];
#pragma unroll
            for (int j = 0; j < 4; ++j) {
                int r = wn * 64 + j * 16 + lr;
                bf[j] = *(const v16h*)&Bs[st][r * 64 + ks * 32 + hs * 16];
            }
#pragma unroll
            for (int i = 0; i < 2; ++i)
#pragma unroll
                for (int j = 0; j < 4; ++j)
                    acc[i][j] = __builtin_amdgcn_wmma_f32_16x16x32_f16(
                        false, af[i], false, bf[j], (short)0, acc[i][j],
                        false, false);
        }
    }

    // ------------------ epilogue ------------------
#pragma unroll
    for (int i = 0; i < 2; ++i) {
#pragma unroll
        for (int j = 0; j < 4; ++j) {
            int gcol = nBase + wn * 64 + j * 16 + lr;
#pragma unroll
            for (int r = 0; r < 8; ++r) {
                int grow = mBase + wm * 32 + i * 16 + r + 8 * hs;
                float val = acc[i][j][r] + bias[gcol];
                if (MODE == 0) {
                    int bb = grow >> 11, ss = grow & 2047;
                    if (gcol < 1024) {
                        int hh = gcol >> 6, d = gcol & 63;
                        qO[((size_t)(bb * 16 + hh) * SEQ + ss) * 64 + d] =
                            (_Float16)(val * 0.125f);
                    } else if (gcol < 2048) {
                        int c2 = gcol - 1024;
                        int hh = c2 >> 6, d = c2 & 63;
                        kO[((size_t)(bb * 16 + hh) * SEQ + ss) * 64 + d] =
                            (_Float16)val;
                    } else {
                        int c2 = gcol - 2048;
                        int hh = c2 >> 6, d = c2 & 63;
                        vTO[((size_t)(bb * 16 + hh) * 64 + d) * SEQ + ss] =
                            (_Float16)val;
                    }
                } else if (MODE == 1) {
                    val += resid[(size_t)grow * N + gcol];
                    outF[(size_t)grow * N + gcol] = val;
                } else if (MODE == 2) {
                    float g = 0.5f * val * (1.0f + erff(val * 0.70710678f));
                    outH[(size_t)grow * N + gcol] = (_Float16)g;
                } else { // MODE == 3
                    val += resid[(size_t)grow * N + gcol];
                    outF[(size_t)grow * N + gcol] = val;
                }
            }
        }
    }
}

// ---------------------------------------------------------------------------
// Flash-style attention.  q pre-scaled by 1/sqrt(64).  v pre-transposed.
// 8 independent waves/block, each owns 16 query rows of one (b,h).
// Key tile = 64.  No block barriers: the LDS P round-trip is wave-private
// and same-wave LDS ops are architecturally in-order.
// grid = B*H*S/16/8 = 1024
// ---------------------------------------------------------------------------
__global__ __launch_bounds__(256)
void attn_kernel(const _Float16* __restrict__ q, const _Float16* __restrict__ k,
                 const _Float16* __restrict__ vT, _Float16* __restrict__ ctx) {
    __shared__ __attribute__((aligned(32))) _Float16 P[8][16 * 64];
    const int tid = threadIdx.x, lane = tid & 31, w = tid >> 5;
    const int t = blockIdx.x * 8 + w;
    const int bh = t >> 7;          // S/16 = 128 m-tiles per (b,h)
    const int qbase = (t & 127) * 16;
    const int lr = lane & 15, hs = lane >> 4;

    const _Float16* qh = q  + (size_t)bh * SEQ * 64;
    const _Float16* kh = k  + (size_t)bh * SEQ * 64;
    const _Float16* vh = vT + (size_t)bh * 64 * SEQ;

    v16h qf[2];
#pragma unroll
    for (int ds = 0; ds < 2; ++ds) {
        size_t base = (size_t)(qbase + lr) * 64 + ds * 32 + hs * 16;
        v8h lo = *(const v8h*)&qh[base];
        v8h hi = *(const v8h*)&qh[base + 8];
        qf[ds] = __builtin_shufflevector(lo, hi, 0,1,2,3,4,5,6,7,
                                                  8,9,10,11,12,13,14,15);
    }

    v8f accv[4];
#pragma unroll
    for (int j = 0; j < 4; ++j) accv[j] = (v8f){0,0,0,0,0,0,0,0};
    float mrow[8], lsum[8];
#pragma unroll
    for (int r = 0; r < 8; ++r) { mrow[r] = -1e30f; lsum[r] = 0.f; }

    for (int kb = 0; kb < SEQ; kb += 64) {
        if (kb + 64 < SEQ) {  // hint next K/V tile toward the caches
            __builtin_prefetch(&kh[(size_t)(kb + 64 + lr) * 64], 0, 0);
            __builtin_prefetch(&vh[(size_t)lr * SEQ + kb + 64], 0, 0);
        }
        // scores: 4 groups of 16 keys, K-dim (head_dim=64) in 2 WMMAs each
        v8f s[4];
#pragma unroll
        for (int g = 0; g < 4; ++g) {
            s[g] = (v8f){0,0,0,0,0,0,0,0};
            v16h kf;
            kf = *(const v16h*)&kh[(size_t)(kb + g * 16 + lr) * 64 + hs * 16];
            s[g] = __builtin_amdgcn_wmma_f32_16x16x32_f16(
                false, qf[0], false, kf, (short)0, s[g], false, false);
            kf = *(const v16h*)&kh[(size_t)(kb + g * 16 + lr) * 64 + 32 + hs * 16];
            s[g] = __builtin_amdgcn_wmma_f32_16x16x32_f16(
                false, qf[1], false, kf, (short)0, s[g], false, false);
        }

        // online softmax over the 64-key slab
#pragma unroll
        for (int r = 0; r < 8; ++r) {
            float vm = fmaxf(fmaxf(s[0][r], s[1][r]), fmaxf(s[2][r], s[3][r]));
#pragma unroll
            for (int msk = 8; msk >= 1; msk >>= 1)
                vm = fmaxf(vm, __shfl_xor(vm, msk, 32));
            float nm = fmaxf(mrow[r], vm);
            float sc = __expf(mrow[r] - nm);
            float pg[4];
            float rs = 0.f;
#pragma unroll
            for (int g = 0; g < 4; ++g) { pg[g] = __expf(s[g][r] - nm); rs += pg[g]; }
#pragma unroll
            for (int msk = 8; msk >= 1; msk >>= 1)
                rs += __shfl_xor(rs, msk, 32);
            lsum[r] = lsum[r] * sc + rs;
            mrow[r] = nm;
            accv[0][r] *= sc; accv[1][r] *= sc;
            accv[2][r] *= sc; accv[3][r] *= sc;
            int prow = r + 8 * hs;
#pragma unroll
            for (int g = 0; g < 4; ++g)
                P[w][prow * 64 + g * 16 + lr] = (_Float16)pg[g];
        }

        // reshape P (C layout -> A layout) through wave-private LDS
        v16h pf[2];
#pragma unroll
        for (int ks = 0; ks < 2; ++ks) {
            v8h plo = *(const v8h*)&P[w][lr * 64 + ks * 32 + hs * 16];
            v8h phi = *(const v8h*)&P[w][lr * 64 + ks * 32 + hs * 16 + 8];
            pf[ks] = __builtin_shufflevector(plo, phi, 0,1,2,3,4,5,6,7,
                                                        8,9,10,11,12,13,14,15);
        }
        // ctx += P @ V  (V pre-transposed: contiguous along keys)
#pragma unroll
        for (int j = 0; j < 4; ++j) {
            v16h vf;
            vf = *(const v16h*)&vh[(size_t)(j * 16 + lr) * SEQ + kb + hs * 16];
            accv[j] = __builtin_amdgcn_wmma_f32_16x16x32_f16(
                false, pf[0], false, vf, (short)0, accv[j], false, false);
            vf = *(const v16h*)&vh[(size_t)(j * 16 + lr) * SEQ + kb + 32 + hs * 16];
            accv[j] = __builtin_amdgcn_wmma_f32_16x16x32_f16(
                false, pf[1], false, vf, (short)0, accv[j], false, false);
        }
    }

    const int bb = bh >> 4, hh = bh & 15;
#pragma unroll
    for (int r = 0; r < 8; ++r) {
        float inv = 1.f / lsum[r];
        int srow = qbase + r + 8 * hs;
#pragma unroll
        for (int j = 0; j < 4; ++j)
            ctx[((size_t)(bb * SEQ + srow)) * D_MODEL + hh * 64 + j * 16 + lr] =
                (_Float16)(accv[j][r] * inv);
    }
}

// ---------------------------------------------------------------------------
// Host launcher
// ---------------------------------------------------------------------------
extern "C" void kernel_launch(void* const* d_in, const int* in_sizes, int n_in,
                              void* d_out, int out_size, void* d_ws, size_t ws_size,
                              hipStream_t stream) {
    const float* src     = (const float*)d_in[0];
    const float* qkv_w   = (const float*)d_in[1];
    const float* qkv_b   = (const float*)d_in[2];
    const float* out_w   = (const float*)d_in[3];
    const float* out_b   = (const float*)d_in[4];
    const float* ff1_w   = (const float*)d_in[5];
    const float* ff1_b   = (const float*)d_in[6];
    const float* ff2_w   = (const float*)d_in[7];
    const float* ff2_b   = (const float*)d_in[8];
    const float* norm1_w = (const float*)d_in[9];
    const float* norm2_w = (const float*)d_in[10];

    char* ws = (char*)d_ws;
    size_t off = 0;
    auto take = [&](size_t bytes) {
        char* p = ws + off;
        off = (off + bytes + 255) & ~(size_t)255;
        return p;
    };
    _Float16* xn    = (_Float16*)take((size_t)M_TOT * D_MODEL * 2);
    _Float16* qkvT  = (_Float16*)take((size_t)D_MODEL * 3 * D_MODEL * 2);
    _Float16* outT  = (_Float16*)take((size_t)D_MODEL * D_MODEL * 2);
    _Float16* ff1T  = (_Float16*)take((size_t)D_MODEL * D_FF * 2);
    _Float16* ff2T  = (_Float16*)take((size_t)D_FF * D_MODEL * 2);
    _Float16* qbuf  = (_Float16*)take((size_t)M_TOT * D_MODEL * 2);
    _Float16* kbuf  = (_Float16*)take((size_t)M_TOT * D_MODEL * 2);
    _Float16* vTbuf = (_Float16*)take((size_t)M_TOT * D_MODEL * 2);
    _Float16* ctx   = (_Float16*)take((size_t)M_TOT * D_MODEL * 2);
    float*    src2  = (float*)   take((size_t)M_TOT * D_MODEL * 4);
    _Float16* ybuf  = (_Float16*)take((size_t)M_TOT * D_MODEL * 2);
    _Float16* hbuf  = (_Float16*)take((size_t)M_TOT * D_FF * 2);
    (void)ws_size; (void)in_sizes; (void)n_in; (void)out_size;

    dim3 tb(32, 8);
    transpose_cvt<<<dim3(3 * D_MODEL / 32, D_MODEL / 32), tb, 0, stream>>>(
        qkv_w, qkvT, D_MODEL, 3 * D_MODEL);
    transpose_cvt<<<dim3(D_MODEL / 32, D_MODEL / 32), tb, 0, stream>>>(
        out_w, outT, D_MODEL, D_MODEL);
    transpose_cvt<<<dim3(D_FF / 32, D_MODEL / 32), tb, 0, stream>>>(
        ff1_w, ff1T, D_MODEL, D_FF);
    transpose_cvt<<<dim3(D_MODEL / 32, D_FF / 32), tb, 0, stream>>>(
        ff2_w, ff2T, D_FF, D_MODEL);

    rmsnorm_kernel<<<M_TOT, 256, 0, stream>>>(src, norm1_w, xn, D_MODEL);

    gemm_f16<0><<<dim3(3 * D_MODEL / 128, M_TOT / 128), 256, 0, stream>>>(
        xn, qkvT, qkv_b, nullptr, nullptr, nullptr, qbuf, kbuf, vTbuf,
        M_TOT, 3 * D_MODEL, D_MODEL);

    attn_kernel<<<BATCH * NHEAD * (SEQ / 16) / 8, 256, 0, stream>>>(
        qbuf, kbuf, vTbuf, ctx);

    gemm_f16<1><<<dim3(D_MODEL / 128, M_TOT / 128), 256, 0, stream>>>(
        ctx, outT, out_b, src, src2, nullptr, nullptr, nullptr, nullptr,
        M_TOT, D_MODEL, D_MODEL);

    rmsnorm_kernel<<<M_TOT, 256, 0, stream>>>(src2, norm2_w, ybuf, D_MODEL);

    gemm_f16<2><<<dim3(D_FF / 128, M_TOT / 128), 256, 0, stream>>>(
        ybuf, ff1T, ff1_b, nullptr, nullptr, hbuf, nullptr, nullptr, nullptr,
        M_TOT, D_FF, D_MODEL);

    gemm_f16<3><<<dim3(D_MODEL / 128, M_TOT / 128), 256, 0, stream>>>(
        hbuf, ff2T, ff2_b, src2, (float*)d_out, nullptr, nullptr, nullptr,
        nullptr, M_TOT, D_MODEL, D_FF);
}
